// MultiHeadSelfAttention_52441550684552
// MI455X (gfx1250) — matrix operand order repaired and checked by reading, not compiled, not run on hardware
//
#include <hip/hip_runtime.h>
#include <stdint.h>

// MultiHeadSelfAttention for MI455X (gfx1250, wave32, WMMA + TDM).
// B=2, S=2048, D=1024, H=16, DK=64.
// All GEMMs use v_wmma_f32_16x16x32_f16 (f16 inputs, f32 accumulate).
// Attention stages K/V tiles into LDS via TENSOR_LOAD_TO_LDS (double-buffered,
// TENSORcnt-tracked) so 8 waves share one copy instead of 8 redundant reads.
// Workspace requirement: ~68 MB.

#define B_  2
#define S_  2048
#define D_  1024
#define H_  16
#define DK_ 64

typedef __attribute__((ext_vector_type(16))) _Float16 v16h;
typedef __attribute__((ext_vector_type(8)))  _Float16 v8h;
typedef __attribute__((ext_vector_type(4)))  _Float16 v4h;
typedef __attribute__((ext_vector_type(8)))  float    v8f;
typedef __attribute__((ext_vector_type(4)))  unsigned v4u;
typedef __attribute__((ext_vector_type(8)))  int      v8i;
typedef __attribute__((ext_vector_type(4)))  int      v4i;

#define WMMA_F16F32(a, b, c) \
  __builtin_amdgcn_wmma_f32_16x16x32_f16(false, (a), false, (b), (short)0, (c), false, false)

#if defined(__gfx1250__) && __has_builtin(__builtin_amdgcn_tensor_load_to_lds)
#define USE_TDM 1
#else
#define USE_TDM 0
#endif

// ---------------------------------------------------------------------------
// Fragment loaders (wave32; this file's contiguous k order, same for A and B)
// A 16x32 f16 (M x K): lane row = lane%16; 16 contiguous halves,
//   K = 0..15 (lanes 0-15) or 16..31 (lanes 16-31).
// B 32x16 f16 (K x N): lane col = lane%16; 16 contiguous halves,
//   K = 0..15 (lanes 0-15) or 16..31 (lanes 16-31).
// C/D 16x16 f32: lane col = lane%16; VGPR r -> row r (+8 for lanes 16-31).
// ---------------------------------------------------------------------------
__device__ __forceinline__ v16h load_a_frag(const _Float16* rowPtr, int kb, int lane) {
  const int hi = (lane >= 16) ? 16 : 0;
  return *(const v16h*)(rowPtr + kb + hi);
}

__device__ __forceinline__ v16h load_b_frag(const _Float16* rowPtr, int kb, int lane) {
  const int hi = (lane >= 16) ? 16 : 0;
  return *(const v16h*)(rowPtr + kb + hi);
}

// ---------------------------------------------------------------------------
// Tensor Data Mover: 2-D tile DMA, global -> LDS (ISA ch.8 D# layout).
// data_size = 2 bytes (f16). Tensor dims == tile dims so no element is OOB.
// Groups 2/3 carry benign dims (tensor_dim2/3/4 = 1, tile_dim3/4 = 0).
// clang-23 builtin arity is 6: (g0 v4u, g1 v8i, g2 v4i, g3 v4i, g4 v8i, cpol);
// the 5th operand maps to the unused VADDR4 group (zeros).
// ---------------------------------------------------------------------------
#if USE_TDM
__device__ __forceinline__ void tdm_load_2d_f16(unsigned lds_off, const _Float16* gaddr,
                                                unsigned tile_x, unsigned tile_y,
                                                unsigned row_stride_elems) {
  const unsigned long long ga = (unsigned long long)(uintptr_t)gaddr;
  v4u g0;
  g0[0] = 1u;                                            // count=1, user desc
  g0[1] = lds_off;                                       // LDS byte address
  g0[2] = (unsigned)(ga & 0xffffffffu);                  // global_addr[31:0]
  g0[3] = (unsigned)((ga >> 32) & 0x01ffffffu) | (2u << 30);  // addr[56:32] | type=2
  v8i g1;
  g1[0] = (int)(1u << 16);                               // wg_mask=0, data_size=1 (2B)
  g1[1] = (int)((tile_x & 0xffffu) << 16);               // tensor_dim0[15:0]
  g1[2] = (int)(((tile_x >> 16) & 0xffffu) | ((tile_y & 0xffffu) << 16));  // td0 hi | td1 lo
  g1[3] = (int)(((tile_y >> 16) & 0xffffu) | ((tile_x & 0xffffu) << 16));  // td1 hi | tile_dim0
  g1[4] = (int)(tile_y & 0xffffu);                       // tile_dim1 | tile_dim2=0
  g1[5] = (int)row_stride_elems;                         // tensor_dim0_stride[31:0]
  g1[6] = 0;                                             // stride0 hi | stride1 lo
  g1[7] = 0;                                             // stride1 hi
  v4i g2;
  g2[0] = 1;                                             // tensor_dim2 = 1
  g2[1] = 1;                                             // tensor_dim3 = 1
  g2[2] = 0;                                             // dim2 stride lo
  g2[3] = 0;                                             // dim2 stride hi | tile_dim3=0
  v4i g3;
  g3[0] = 0;                                             // dim3 stride lo
  g3[1] = (int)(1u << 16);                               // dim3 stride hi | tensor_dim4=1
  g3[2] = 0;                                             // td4 hi | tile_dim4=0
  g3[3] = 0;
  v8i g4 = {0, 0, 0, 0, 0, 0, 0, 0};                     // VADDR4: unused
  __builtin_amdgcn_tensor_load_to_lds(g0, g1, g2, g3, g4, 0);
}
#endif

__device__ __forceinline__ void wait_tensorcnt0() {
#if __has_builtin(__builtin_amdgcn_s_wait_tensorcnt)
  __builtin_amdgcn_s_wait_tensorcnt(0);
#else
  asm volatile("s_wait_tensorcnt 0x0" ::: "memory");
#endif
}

// ---------------------------------------------------------------------------
// f32 -> f16 convert, 4 elements per thread
// ---------------------------------------------------------------------------
__global__ void cvt_f32_to_f16(const float* __restrict__ in, _Float16* __restrict__ out, int n4) {
  int i = blockIdx.x * blockDim.x + threadIdx.x;
  if (i >= n4) return;
  float4 f = ((const float4*)in)[i];
  v4h h = { (_Float16)f.x, (_Float16)f.y, (_Float16)f.z, (_Float16)f.w };
  ((v4h*)out)[i] = h;
}

// ---------------------------------------------------------------------------
// Pack int32 mask [B,1,S,S] into bitmask: 1 dword per (b, q, 32 keys).
// Kills the 16x per-head re-read of the 33.5 MB int mask (would be 537 MB).
// ---------------------------------------------------------------------------
__global__ void pack_mask_bits(const int* __restrict__ mask, unsigned* __restrict__ bits, int nwords) {
  int i = blockIdx.x * blockDim.x + threadIdx.x;
  if (i >= nwords) return;
  const int* p = mask + (size_t)i * 32;
  unsigned w = 0;
#pragma unroll
  for (int j = 0; j < 32; ++j) w |= (p[j] != 0 ? 1u : 0u) << j;
  bits[i] = w;
}

// ---------------------------------------------------------------------------
// GEMM: C[M,N] = A[M,K] @ W[N,K]^T (nn.Linear layout; both f16 row-major).
// Block: 256 threads = 8 waves; block tile 128(M) x 64(N); wave tile 32x32.
// mode 0: write f32 row-major [M,N] (final Wo projection -> d_out)
// mode 1: write f16 head-split [B,H,S,DK]   (Q/K projections)
// mode 2: write f16 head-split-T [B,H,DK,S] (V projection)
// ---------------------------------------------------------------------------
__global__ __launch_bounds__(256) void
gemm_xWT_wmma(const _Float16* __restrict__ A, const _Float16* __restrict__ W,
              float* __restrict__ outF, _Float16* __restrict__ outH,
              int M, int N, int K, int mode) {
  const int lane = threadIdx.x & 31;
  const int wave = threadIdx.x >> 5;
  const int wm = wave & 3;        // 4 waves along M
  const int wn = wave >> 2;       // 2 waves along N
  const int l16 = lane & 15;
  const int hi8 = (lane >= 16) ? 8 : 0;
  const int mBase = blockIdx.x * 128 + wm * 32;
  const int nBase = blockIdx.y * 64 + wn * 32;

  const _Float16* aRow0 = A + (size_t)(mBase + l16) * K;
  const _Float16* aRow1 = aRow0 + (size_t)16 * K;
  const _Float16* bRow0 = W + (size_t)(nBase + l16) * K;
  const _Float16* bRow1 = bRow0 + (size_t)16 * K;

  const v8f vz = {0.f, 0.f, 0.f, 0.f, 0.f, 0.f, 0.f, 0.f};
  v8f acc00 = vz, acc01 = vz, acc10 = vz, acc11 = vz;

  for (int kb = 0; kb < K; kb += 32) {
    v16h a0 = load_a_frag(aRow0, kb, lane);
    v16h a1 = load_a_frag(aRow1, kb, lane);
    v16h b0 = load_b_frag(bRow0, kb, lane);
    v16h b1 = load_b_frag(bRow1, kb, lane);
    acc00 = WMMA_F16F32(a0, b0, acc00);
    acc01 = WMMA_F16F32(a0, b1, acc01);
    acc10 = WMMA_F16F32(a1, b0, acc10);
    acc11 = WMMA_F16F32(a1, b1, acc11);
  }

  const v8f* accs[4] = {&acc00, &acc01, &acc10, &acc11};
#pragma unroll
  for (int f = 0; f < 4; ++f) {
    const int mi = f >> 1, ni = f & 1;
    const int n = nBase + ni * 16 + l16;
#pragma unroll
    for (int r = 0; r < 8; ++r) {
      const int m = mBase + mi * 16 + hi8 + r;
      const float v = (*accs[f])[r];
      if (mode == 0) {
        outF[(size_t)m * N + n] = v;
      } else {
        const int b = m >> 11;              // m / S_
        const int s = m & (S_ - 1);
        const int h = n >> 6;               // n / DK_
        const int dk = n & (DK_ - 1);
        const size_t idx = (mode == 1)
            ? ((((size_t)b * H_ + h) * S_ + s) * DK_ + dk)
            : ((((size_t)b * H_ + h) * DK_ + dk) * S_ + s);
        outH[idx] = (_Float16)v;
      }
    }
  }
}

// ---------------------------------------------------------------------------
// Flash attention, one (b,h) per blockIdx.y, 128 query rows per block,
// 16 query rows per wave. K/V tiles staged into LDS (TDM double-buffered;
// one copy shared by 8 waves). Online softmax in f32; P staged through a
// per-wave LDS tile (C-layout -> A-fragment layout); PV uses transposed Vt.
// ---------------------------------------------------------------------------
__global__ __launch_bounds__(256) void
flash_attn_wmma(const _Float16* __restrict__ Qh, const _Float16* __restrict__ Kh,
                const _Float16* __restrict__ Vt, const unsigned* __restrict__ maskBits,
                _Float16* __restrict__ ctx) {
  __shared__ _Float16 tileK[2][32][64];  // 2 x 4 KB: 32 keys x DK=64
  __shared__ _Float16 tileV[2][64][32];  // 2 x 4 KB: DK=64 x 32 keys (from Vt)
  __shared__ _Float16 ldsP[8][16][32];   // per-wave 16x32 P tile (8 KB)

  const int lane = threadIdx.x & 31;
  const int wave = threadIdx.x >> 5;
  const int l16 = lane & 15;
  const int hi8 = (lane >= 16) ? 8 : 0;
  const int bh = blockIdx.y;            // 0..31
  const int b = bh >> 4;
  const int h = bh & (H_ - 1);
  const int qBase = blockIdx.x * 128 + wave * 16;

  const _Float16* Ksrc = Kh + (size_t)bh * S_ * DK_;   // [S, DK] row-major
  const _Float16* Vsrc = Vt + (size_t)bh * DK_ * S_;   // [DK, S] row-major

  // Q A-fragments for this wave's 16 query rows (DK=64 -> two 32-K chunks)
  const _Float16* Qrow = Qh + ((size_t)bh * S_ + qBase + l16) * DK_;
  v16h aQ0 = load_a_frag(Qrow, 0, lane);
  v16h aQ1 = load_a_frag(Qrow, 32, lane);

  const v8f vz = {0.f, 0.f, 0.f, 0.f, 0.f, 0.f, 0.f, 0.f};
  v8f o0 = vz, o1 = vz, o2 = vz, o3 = vz;   // ctx accum: 4 dk tiles of 16
  float mrow[8], lrow[8];
#pragma unroll
  for (int r = 0; r < 8; ++r) { mrow[r] = -1e30f; lrow[r] = 0.f; }

  const unsigned* mbase = maskBits + (size_t)b * S_ * (S_ / 32);

#if USE_TDM
  if (wave == 0) {  // prime buffer 0 (chunk 0)
    tdm_load_2d_f16((unsigned)(uintptr_t)&tileK[0][0][0], Ksrc, 64u, 32u, 64u);
    tdm_load_2d_f16((unsigned)(uintptr_t)&tileV[0][0][0], Vsrc, 32u, 64u, (unsigned)S_);
  }
#endif

  for (int kb = 0; kb < S_; kb += 32) {
    const int buf = (kb >> 5) & 1;
#if USE_TDM
    if (wave == 0) wait_tensorcnt0();     // chunk kb landed in LDS
    __syncthreads();                      // ...and everyone finished buf^1 reads
    if (wave == 0 && (kb + 32) < S_) {    // kick DMA for next chunk into buf^1
      tdm_load_2d_f16((unsigned)(uintptr_t)&tileK[buf ^ 1][0][0],
                      Ksrc + (size_t)(kb + 32) * DK_, 64u, 32u, 64u);
      tdm_load_2d_f16((unsigned)(uintptr_t)&tileV[buf ^ 1][0][0],
                      Vsrc + (kb + 32), 32u, 64u, (unsigned)S_);
    }
#else
    {
      const int t = threadIdx.x;          // cooperative fallback: 256 x 16B each
      *(v8h*)&tileK[buf][t >> 3][(t & 7) * 8] =
          *(const v8h*)(Ksrc + (size_t)(kb + (t >> 3)) * DK_ + (t & 7) * 8);
      *(v8h*)&tileV[buf][t >> 2][(t & 3) * 8] =
          *(const v8h*)(Vsrc + (size_t)(t >> 2) * S_ + kb + (t & 3) * 8);
    }
    __syncthreads();
#endif

    // ---- scores: two 16-key subtiles from LDS K tile (2 WMMAs each, K=64)
    v8f s0 = vz, s1 = vz;
    {
      const _Float16* kRow0 = &tileK[buf][l16][0];
      const _Float16* kRow1 = &tileK[buf][16 + l16][0];
      v16h kb00 = load_b_frag(kRow0, 0, lane);
      v16h kb01 = load_b_frag(kRow0, 32, lane);
      v16h kb10 = load_b_frag(kRow1, 0, lane);
      v16h kb11 = load_b_frag(kRow1, 32, lane);
      s0 = WMMA_F16F32(aQ0, kb00, s0);
      s0 = WMMA_F16F32(aQ1, kb01, s0);
      s1 = WMMA_F16F32(aQ0, kb10, s1);
      s1 = WMMA_F16F32(aQ1, kb11, s1);
    }

    // ---- scale + mask + running row max (16-lane shuffle reduction)
    float mnew[8];
#pragma unroll
    for (int r = 0; r < 8; ++r) {
      const int qrow = qBase + hi8 + r;
      const unsigned mw = mbase[(size_t)qrow * (S_ / 32) + (kb >> 5)];
      float v0 = ((mw >> l16) & 1u)        ? s0[r] * 0.125f : -1e9f;
      float v1 = ((mw >> (16 + l16)) & 1u) ? s1[r] * 0.125f : -1e9f;
      s0[r] = v0; s1[r] = v1;
      float mx = fmaxf(v0, v1);
      mx = fmaxf(mx, __shfl_xor(mx, 1));
      mx = fmaxf(mx, __shfl_xor(mx, 2));
      mx = fmaxf(mx, __shfl_xor(mx, 4));
      mx = fmaxf(mx, __shfl_xor(mx, 8));
      mnew[r] = fmaxf(mrow[r], mx);
    }

    // ---- exp, running sum, rescale accumulators
#pragma unroll
    for (int r = 0; r < 8; ++r) {
      float p0 = __expf(s0[r] - mnew[r]);
      float p1 = __expf(s1[r] - mnew[r]);
      float sum = p0 + p1;
      sum += __shfl_xor(sum, 1);
      sum += __shfl_xor(sum, 2);
      sum += __shfl_xor(sum, 4);
      sum += __shfl_xor(sum, 8);
      const float alpha = __expf(mrow[r] - mnew[r]);
      lrow[r] = lrow[r] * alpha + sum;
      mrow[r] = mnew[r];
      s0[r] = p0; s1[r] = p1;
      o0[r] *= alpha; o1[r] *= alpha; o2[r] *= alpha; o3[r] *= alpha;
    }

    // ---- P: C-layout -> f16 A-fragment layout via per-wave LDS tile
#pragma unroll
    for (int r = 0; r < 8; ++r) {
      ldsP[wave][hi8 + r][l16]      = (_Float16)s0[r];
      ldsP[wave][hi8 + r][16 + l16] = (_Float16)s1[r];
    }
    asm volatile("s_wait_dscnt 0x0" ::: "memory");  // same-wave LDS RAW
    v16h aP = load_a_frag(&ldsP[wave][l16][0], 0, lane);

    // ---- ctx += P @ V : B-fragments are contiguous rows of the LDS V tile
    o0 = WMMA_F16F32(aP, load_b_frag(&tileV[buf][ 0 + l16][0], 0, lane), o0);
    o1 = WMMA_F16F32(aP, load_b_frag(&tileV[buf][16 + l16][0], 0, lane), o1);
    o2 = WMMA_F16F32(aP, load_b_frag(&tileV[buf][32 + l16][0], 0, lane), o2);
    o3 = WMMA_F16F32(aP, load_b_frag(&tileV[buf][48 + l16][0], 0, lane), o3);

#if !USE_TDM
    __syncthreads();   // reads done before next chunk overwrites the buffer
#endif
  }

  // ---- normalize and write ctx back in [B,S,D] f16 for the Wo GEMM
#pragma unroll
  for (int r = 0; r < 8; ++r) {
    const int qrow = qBase + hi8 + r;
    _Float16* orow = ctx + ((size_t)b * S_ + qrow) * D_ + h * DK_;
    const float inv = 1.0f / lrow[r];
    orow[ 0 + l16] = (_Float16)(o0[r] * inv);
    orow[16 + l16] = (_Float16)(o1[r] * inv);
    orow[32 + l16] = (_Float16)(o2[r] * inv);
    orow[48 + l16] = (_Float16)(o3[r] * inv);
  }
}

// ---------------------------------------------------------------------------
extern "C" void kernel_launch(void* const* d_in, const int* in_sizes, int n_in,
                              void* d_out, int out_size, void* d_ws, size_t ws_size,
                              hipStream_t stream) {
  (void)in_sizes; (void)n_in; (void)out_size; (void)ws_size;
  const float* q    = (const float*)d_in[0];
  const float* k    = (const float*)d_in[1];
  const float* v    = (const float*)d_in[2];
  const int*   mask = (const int*)d_in[3];
  const float* Wq   = (const float*)d_in[4];
  const float* Wk   = (const float*)d_in[5];
  const float* Wv   = (const float*)d_in[6];
  const float* Wo   = (const float*)d_in[7];
  float* out = (float*)d_out;

  const size_t NBSD = (size_t)B_ * S_ * D_;  // 4,194,304
  const size_t NDD  = (size_t)D_ * D_;       // 1,048,576
  _Float16* q16   = (_Float16*)d_ws;
  _Float16* k16   = q16 + NBSD;
  _Float16* v16   = k16 + NBSD;
  _Float16* wq16  = v16 + NBSD;
  _Float16* wk16  = wq16 + NDD;
  _Float16* wv16  = wk16 + NDD;
  _Float16* wo16  = wv16 + NDD;
  _Float16* Qh    = wo16 + NDD;              // [B,H,S,DK]
  _Float16* Kh    = Qh + NBSD;               // [B,H,S,DK]
  _Float16* Vt    = Kh + NBSD;               // [B,H,DK,S]
  _Float16* ctx16 = Vt + NBSD;               // [B,S,D]
  unsigned* maskBits = (unsigned*)(ctx16 + NBSD);  // B*S*S/32 dwords (1 MB)

  // 1) precision convert (f32 -> f16)
  {
    const int n4 = (int)(NBSD / 4);          // 1,048,576
    cvt_f32_to_f16<<<n4 / 256, 256, 0, stream>>>(q, q16, n4);
    cvt_f32_to_f16<<<n4 / 256, 256, 0, stream>>>(k, k16, n4);
    cvt_f32_to_f16<<<n4 / 256, 256, 0, stream>>>(v, v16, n4);
    const int w4 = (int)(NDD / 4);           // 262,144
    cvt_f32_to_f16<<<w4 / 256, 256, 0, stream>>>(Wq, wq16, w4);
    cvt_f32_to_f16<<<w4 / 256, 256, 0, stream>>>(Wk, wk16, w4);
    cvt_f32_to_f16<<<w4 / 256, 256, 0, stream>>>(Wv, wv16, w4);
    cvt_f32_to_f16<<<w4 / 256, 256, 0, stream>>>(Wo, wo16, w4);
  }

  // 2) pack mask into bits
  {
    const int nwords = B_ * S_ * S_ / 32;    // 262,144
    pack_mask_bits<<<nwords / 256, 256, 0, stream>>>(mask, maskBits, nwords);
  }

  // 3) Q/K/V projections (WMMA GEMM, head-split outputs; V transposed)
  const dim3 gg((B_ * S_) / 128, D_ / 64);   // (32, 16)
  gemm_xWT_wmma<<<gg, 256, 0, stream>>>(q16, wq16, nullptr, Qh, B_ * S_, D_, D_, 1);
  gemm_xWT_wmma<<<gg, 256, 0, stream>>>(k16, wk16, nullptr, Kh, B_ * S_, D_, D_, 1);
  gemm_xWT_wmma<<<gg, 256, 0, stream>>>(v16, wv16, nullptr, Vt, B_ * S_, D_, D_, 2);

  // 4) masked flash attention (TDM-staged K/V tiles)
  flash_attn_wmma<<<dim3(S_ / 128, B_ * H_), 256, 0, stream>>>(Qh, Kh, Vt, maskBits, ctx16);

  // 5) output projection -> f32
  gemm_xWT_wmma<<<gg, 256, 0, stream>>>(ctx16, wo16, out, nullptr, B_ * S_, D_, D_, 0);
}
